// Attention2Conv_10797547782216
// MI455X (gfx1250) — matrix operand order repaired and checked by reading, not compiled, not run on hardware
//
#include <hip/hip_runtime.h>

typedef __attribute__((ext_vector_type(16))) _Float16 v16h;
typedef __attribute__((ext_vector_type(8)))  float    v8f;

#define HDIM 128
#define BN_EPS 1e-5f

// ---------------------------------------------------------------- utilities
__global__ void fill_kernel(float* p, float v, long n) {
  long i = (long)blockIdx.x * blockDim.x + threadIdx.x;
  if (i < n) p[i] = v;
}

__global__ void deg_accum_kernel(const int* __restrict__ dst, float* deg, int nE) {
  int e = blockIdx.x * blockDim.x + threadIdx.x;
  if (e < nE) unsafeAtomicAdd(&deg[dst[e]], 1.0f);
}

__global__ void rsqrt_kernel(float* p, int n) {
  int i = blockIdx.x * blockDim.x + threadIdx.x;
  if (i < n) { float d = p[i]; p[i] = (d > 0.f) ? rsqrtf(d) : 0.f; }
}

// ---------------------------------------------------------------- WMMA GEMM
// C[n,128] = A[n,128] @ B[128,128], fp32 in/out, fp16 multiply / fp32 accumulate.
// 256 threads = 8 waves; each wave computes a 16-row x 128-col strip.
// B is staged transposed in LDS as f16 so each 16x32x16 B-fragment is two
// contiguous ds_load_b128 per lane.
__global__ __launch_bounds__(256) void gemm_wmma_kernel(const float* __restrict__ A,
    const float* __restrict__ B, float* __restrict__ C, int n) {
  __shared__ _Float16 Bsh[HDIM * HDIM];   // Bsh[col*128 + k] (transposed), 32 KB

  const int tid = threadIdx.x;
#pragma unroll
  for (int it = 0; it < (HDIM * HDIM) / 256; ++it) {
    int i = tid + it * 256;
    int k = i >> 7, c = i & 127;
    Bsh[c * HDIM + k] = (_Float16)B[i];
  }
  __syncthreads();

  const int lane = tid & 31;
  const int wave = tid >> 5;
  const int tile = blockIdx.x * 8 + wave;        // 16-row output tile index
  const int ntiles = (n + 15) >> 4;
  const bool valid = tile < ntiles;
  const int r  = lane & 15;                      // row-in-tile / col-in-tile
  const int kg = lane >> 4;                      // K half selector

  int arow = tile * 16 + r;
  if (arow >= n) arow = n - 1;                   // clamp loads; stores guarded

  v8f zero = {};
  v8f acc[8];
#pragma unroll
  for (int t = 0; t < 8; ++t) acc[t] = zero;

  if (valid) {
    const float* Arow = A + (long)arow * HDIM;
#pragma unroll
    for (int kt = 0; kt < 4; ++kt) {
      // A fragment: 16x32 f16. lane holds row r; halves 0..15 = K kt*32 + kg*16 .. +15
      // (one contiguous 64B run).
      v16h a;
      {
        const float4* p0 = (const float4*)(Arow + kt * 32 + kg * 16);
        const float4* p1 = (const float4*)(Arow + kt * 32 + kg * 16 + 8);
        float4 f0 = p0[0], f1 = p0[1], f2 = p1[0], f3 = p1[1];
        a[0]=(_Float16)f0.x;  a[1]=(_Float16)f0.y;  a[2]=(_Float16)f0.z;  a[3]=(_Float16)f0.w;
        a[4]=(_Float16)f1.x;  a[5]=(_Float16)f1.y;  a[6]=(_Float16)f1.z;  a[7]=(_Float16)f1.w;
        a[8]=(_Float16)f2.x;  a[9]=(_Float16)f2.y;  a[10]=(_Float16)f2.z; a[11]=(_Float16)f2.w;
        a[12]=(_Float16)f3.x; a[13]=(_Float16)f3.y; a[14]=(_Float16)f3.z; a[15]=(_Float16)f3.w;
      }
#pragma unroll
      for (int nt = 0; nt < 8; ++nt) {
        // B fragment: 32x16 f16. lane holds col r; K run = kt*32 + kg*16 .. +15
        // -> 16 contiguous halves in transposed LDS image (32B aligned).
        const int col = nt * 16 + r;
        const v16h* bp = (const v16h*)(Bsh + col * HDIM + kt * 32 + kg * 16);
        v16h b = *bp;
        acc[nt] = __builtin_amdgcn_wmma_f32_16x16x32_f16(
            false, a, false, b, (short)0, acc[nt], false, false);
      }
    }
    // C/D layout: VGPR rr, lanes 0-15 -> M=rr, lanes 16-31 -> M=rr+8; N = r.
#pragma unroll
    for (int nt = 0; nt < 8; ++nt) {
#pragma unroll
      for (int rr = 0; rr < 8; ++rr) {
        int orow = tile * 16 + rr + kg * 8;
        if (orow < n) C[(long)orow * HDIM + nt * 16 + r] = acc[nt][rr];
      }
    }
  }
}

// ---------------------------------------------------------------- edge scatter
// One wave per edge: gather float4 of hw[src], scale by dis[s]*dis[d],
// scatter-add into out[dst] with hardware f32 atomics.
__global__ __launch_bounds__(256) void edge_agg_kernel(const float* __restrict__ hw,
    const float* __restrict__ dis, const int* __restrict__ src,
    const int* __restrict__ dst, float* out, int nE) {
  int e = blockIdx.x * 8 + (threadIdx.x >> 5);
  if (e >= nE) return;
  int lane = threadIdx.x & 31;
  int s = src[e], d = dst[e];
  float nrm = dis[s] * dis[d];
  float4 v = ((const float4*)(hw + (long)s * HDIM))[lane];
  float* o = out + (long)d * HDIM + lane * 4;
  unsafeAtomicAdd(o + 0, v.x * nrm);
  unsafeAtomicAdd(o + 1, v.y * nrm);
  unsafeAtomicAdd(o + 2, v.z * nrm);
  unsafeAtomicAdd(o + 3, v.w * nrm);
}

// ---------------------------------------------------------------- fused epilogue
// h = relu(bn(agg + hw*dis^2 + bias)), written in place over agg.
__global__ __launch_bounds__(256) void post_bn_relu_kernel(float* agg,
    const float* __restrict__ hw, const float* __restrict__ dis,
    const float* __restrict__ bias, const float* __restrict__ gamma,
    const float* __restrict__ beta, const float* __restrict__ mean,
    const float* __restrict__ var, int n) {
  long idx = (long)blockIdx.x * blockDim.x + threadIdx.x;
  long node = idx >> 5;
  if (node >= n) return;
  int lane = (int)(idx & 31);
  int c = lane * 4;
  float di = dis[node];
  float self = di * di;
  float4 a = ((const float4*)(agg + node * HDIM))[lane];
  float4 w = ((const float4*)(hw  + node * HDIM))[lane];
  float4 o;
  float t;
  t = a.x + w.x * self + bias[c+0]; t = (t - mean[c+0]) * gamma[c+0] * rsqrtf(var[c+0] + BN_EPS) + beta[c+0]; o.x = fmaxf(t, 0.f);
  t = a.y + w.y * self + bias[c+1]; t = (t - mean[c+1]) * gamma[c+1] * rsqrtf(var[c+1] + BN_EPS) + beta[c+1]; o.y = fmaxf(t, 0.f);
  t = a.z + w.z * self + bias[c+2]; t = (t - mean[c+2]) * gamma[c+2] * rsqrtf(var[c+2] + BN_EPS) + beta[c+2]; o.z = fmaxf(t, 0.f);
  t = a.w + w.w * self + bias[c+3]; t = (t - mean[c+3]) * gamma[c+3] * rsqrtf(var[c+3] + BN_EPS) + beta[c+3]; o.w = fmaxf(t, 0.f);
  ((float4*)(agg + node * HDIM))[lane] = o;
}

// ---------------------------------------------------------------- attention pool
// One wave per node: att = sigmoid(h . att_w) (wave32 shuffle reduce),
// write att, then atomically pool h*att into g[batch[node]].
__global__ __launch_bounds__(256) void att_pool_kernel(const float* __restrict__ h,
    const float* __restrict__ att_w, const int* __restrict__ batch,
    float* att_out, float* g, int n) {
  long idx = (long)blockIdx.x * blockDim.x + threadIdx.x;
  long node = idx >> 5;
  if (node >= n) return;
  int lane = (int)(idx & 31);
  int c = lane * 4;
  float4 hv = ((const float4*)(h + node * HDIM))[lane];
  float4 aw = ((const float4*)att_w)[lane];
  float s = hv.x * aw.x + hv.y * aw.y + hv.z * aw.z + hv.w * aw.w;
#pragma unroll
  for (int m = 16; m > 0; m >>= 1) s += __shfl_xor(s, m, 32);
  float att = 1.f / (1.f + __expf(-s));
  if (lane == 0) att_out[node] = att;
  int b = batch[node];
  float* gp = g + (long)b * HDIM + c;
  unsafeAtomicAdd(gp + 0, hv.x * att);
  unsafeAtomicAdd(gp + 1, hv.y * att);
  unsafeAtomicAdd(gp + 2, hv.z * att);
  unsafeAtomicAdd(gp + 3, hv.w * att);
}

// ---------------------------------------------------------------- classifier
__global__ void logits_kernel(const float* __restrict__ g,
    const float* __restrict__ cls_w, const float* __restrict__ cls_b,
    float* out) {
  int graph = blockIdx.x;
  int lane = threadIdx.x;
  float4 gv = ((const float4*)(g + (long)graph * HDIM))[lane];
  float4 wv = ((const float4*)cls_w)[lane];
  float s = gv.x * wv.x + gv.y * wv.y + gv.z * wv.z + gv.w * wv.w;
#pragma unroll
  for (int m = 16; m > 0; m >>= 1) s += __shfl_xor(s, m, 32);
  if (lane == 0) out[graph] = s + cls_b[0];
}

// ---------------------------------------------------------------- launch
extern "C" void kernel_launch(void* const* d_in, const int* in_sizes, int n_in,
                              void* d_out, int out_size, void* d_ws, size_t ws_size,
                              hipStream_t stream) {
  const float* x     = (const float*)d_in[0];
  const int*   ei    = (const int*)  d_in[1];
  const int*   batch = (const int*)  d_in[2];
  const float* W1    = (const float*)d_in[3];
  const float* b1    = (const float*)d_in[4];
  const float* g1    = (const float*)d_in[5];
  const float* be1   = (const float*)d_in[6];
  const float* m1    = (const float*)d_in[7];
  const float* v1    = (const float*)d_in[8];
  const float* W2    = (const float*)d_in[9];
  const float* b2    = (const float*)d_in[10];
  const float* g2    = (const float*)d_in[11];
  const float* be2   = (const float*)d_in[12];
  const float* m2    = (const float*)d_in[13];
  const float* v2    = (const float*)d_in[14];
  const float* att_w = (const float*)d_in[15];
  const float* cls_w = (const float*)d_in[16];
  const float* cls_b = (const float*)d_in[17];

  const int n  = in_sizes[0] / HDIM;   // 50000
  const int nE = in_sizes[1] / 2;      // 1600000
  const int G  = out_size - n;         // 64
  const int* src = ei;
  const int* dst = ei + nE;

  // workspace layout (floats): dis[n] | P[n*H] | Q[n*H] | g[G*H]
  float* ws   = (float*)d_ws;
  float* dis  = ws;
  float* P    = ws + n;                  // hw buffer (layer GEMM output)
  float* Q    = P + (long)n * HDIM;      // agg / h buffer
  float* gbuf = Q + (long)n * HDIM;

  float* logits_out = (float*)d_out;       // [0, G)
  float* att_out    = (float*)d_out + G;   // [G, G+n)

  const long nh = (long)n * HDIM;
  const int ntiles  = (n + 15) / 16;
  const int gblocks = (ntiles + 7) / 8;
  const int eblocks = (nE + 7) / 8;
  const int nwaves_blocks = (int)(((long)n * 32 + 255) / 256);

  // degrees (self-loop included) -> dis = rsqrt(deg)
  fill_kernel<<<(n + 255) / 256, 256, 0, stream>>>(dis, 1.0f, n);
  deg_accum_kernel<<<(nE + 255) / 256, 256, 0, stream>>>(dst, dis, nE);
  rsqrt_kernel<<<(n + 255) / 256, 256, 0, stream>>>(dis, n);

  // ---- layer 1
  gemm_wmma_kernel<<<gblocks, 256, 0, stream>>>(x, W1, P, n);
  fill_kernel<<<(int)((nh + 255) / 256), 256, 0, stream>>>(Q, 0.f, nh);
  edge_agg_kernel<<<eblocks, 256, 0, stream>>>(P, dis, src, dst, Q, nE);
  post_bn_relu_kernel<<<nwaves_blocks, 256, 0, stream>>>(Q, P, dis, b1, g1, be1, m1, v1, n);

  // ---- layer 2 (P/Q ping-pong: h1 lives in Q, h2w -> P, agg2 -> Q in place)
  gemm_wmma_kernel<<<gblocks, 256, 0, stream>>>(Q, W2, P, n);
  fill_kernel<<<(int)((nh + 255) / 256), 256, 0, stream>>>(Q, 0.f, nh);
  edge_agg_kernel<<<eblocks, 256, 0, stream>>>(P, dis, src, dst, Q, nE);
  post_bn_relu_kernel<<<nwaves_blocks, 256, 0, stream>>>(Q, P, dis, b2, g2, be2, m2, v2, n);

  // ---- attention pooling + classifier
  fill_kernel<<<((G * HDIM) + 255) / 256, 256, 0, stream>>>(gbuf, 0.f, (long)G * HDIM);
  att_pool_kernel<<<nwaves_blocks, 256, 0, stream>>>(Q, att_w, batch, att_out, gbuf, n);
  logits_kernel<<<G, 32, 0, stream>>>(gbuf, cls_w, cls_b, logits_out);
}
